// GPT_84456236909183
// MI455X (gfx1250) — matrix operand order repaired and checked by reading, not compiled, not run on hardware
//
#include <hip/hip_runtime.h>
#include <hip/hip_bf16.h>
#include <math.h>

#define LNUM 4
#define NS 2
#define NM 2
#define DDIM 768
#define VDIM 50304
#define BDIM 2
#define TDIM 1024
#define BT (BDIM * TDIM)

typedef _Float16 v16h __attribute__((ext_vector_type(16)));
typedef _Float16 v8h  __attribute__((ext_vector_type(8)));
typedef float    v8f  __attribute__((ext_vector_type(8)));

// LDS row stride in halves: 64 data + 8 pad = 72 (144B: 16B-aligned rows,
// 36-dword bank step -> 16 consecutive rows hit 16 distinct banks)
#define LDS_STRIDE 72

// ---------------------------------------------------------------------------
// Tiled WMMA GEMM: C[M,N] = alpha * A[M,K] @ B + bias (+ C if accumulate)
// BTRANS=false: B is [K,N] row-major;  BTRANS=true: B is [N,K] (C = A @ B^T)
// NTSTORE: use non-temporal stores for C (streamed-once outputs, e.g. logits)
// Block = 256 threads (8 waves). Block tile 64(M) x 128(N), K staged by 64,
// LDS double-buffered (1 barrier per stage). Wave tile 32x32: 2 A fragments
// x 2 B fragments x 2 k-steps -> 8 v_wmma per wave per stage, 4 accumulators.
// ---------------------------------------------------------------------------
template <bool BTRANS, bool NTSTORE>
__global__ void __launch_bounds__(256)
gemm_wmma_f16(const float* __restrict__ A, const float* __restrict__ B,
              const float* __restrict__ bias, float* __restrict__ C,
              int M, int N, int K, int lda, int ldb, int ldc,
              float alpha, int accumulate)
{
    __shared__ _Float16 lds_a[2][64  * LDS_STRIDE];
    __shared__ _Float16 lds_b[2][128 * LDS_STRIDE];   // stored as [n][k]

    const int t  = threadIdx.x;
    const int n0 = blockIdx.x * 128;
    const int m0 = blockIdx.y * 64;

    const int lane = t & 31;
    const int wave = t >> 5;
    const int wm   = (wave & 1) * 32;   // 2 waves along M, 32 rows each
    const int wn   = (wave >> 1) * 32;  // 4 waves along N, 32 cols each
    const int lh   = lane & 15;
    const bool hi  = lane >= 16;

    v8f acc[2][2] = {};

    // cooperative-load coordinates
    const int ar = t >> 2;          // 0..63   (A row; 4 float4 along K)
    const int ac = (t & 3) * 16;    // 0,16,32,48
    const int bkr = t >> 2;         // NN: 0..63 (K row; 8 float4 along N)
    const int bnc = (t & 3) * 32;   // NN: 0,32,64,96
    const int bnr = t >> 1;         // NT: 0..127 (N row; 8 float4 along K)
    const int bkc = (t & 1) * 32;   // NT: 0,32

    const int nstage = K >> 6;

    float4 areg[4], breg[8];

    auto load_stage = [&](int st) {
        const int k0 = st << 6;
        const float* ag = A + (size_t)(m0 + ar) * lda + (k0 + ac);
#pragma unroll
        for (int i = 0; i < 4; ++i) areg[i] = *(const float4*)(ag + 4 * i);
        if (BTRANS) {
            const float* bg = B + (size_t)(n0 + bnr) * ldb + (k0 + bkc);
#pragma unroll
            for (int i = 0; i < 8; ++i) breg[i] = *(const float4*)(bg + 4 * i);
        } else {
            const float* bg = B + (size_t)(k0 + bkr) * ldb + (n0 + bnc);
#pragma unroll
            for (int i = 0; i < 8; ++i) breg[i] = *(const float4*)(bg + 4 * i);
        }
    };

    auto store_stage = [&](int buf) {
        _Float16* ad = &lds_a[buf][ar * LDS_STRIDE + ac];
#pragma unroll
        for (int i = 0; i < 4; ++i) {
            ad[4 * i + 0] = (_Float16)areg[i].x;
            ad[4 * i + 1] = (_Float16)areg[i].y;
            ad[4 * i + 2] = (_Float16)areg[i].z;
            ad[4 * i + 3] = (_Float16)areg[i].w;
        }
        if (BTRANS) {
            // breg holds B[n0+bnr][k0+bkc .. +31] -> contiguous into lds_b[n][k]
            _Float16* bd = &lds_b[buf][bnr * LDS_STRIDE + bkc];
#pragma unroll
            for (int i = 0; i < 8; ++i) {
                bd[4 * i + 0] = (_Float16)breg[i].x;
                bd[4 * i + 1] = (_Float16)breg[i].y;
                bd[4 * i + 2] = (_Float16)breg[i].z;
                bd[4 * i + 3] = (_Float16)breg[i].w;
            }
        } else {
            // breg holds B[k0+bkr][n0+bnc .. +31] -> transpose into lds_b[n][k]
#pragma unroll
            for (int i = 0; i < 8; ++i) {
                lds_b[buf][(bnc + 4 * i + 0) * LDS_STRIDE + bkr] = (_Float16)breg[i].x;
                lds_b[buf][(bnc + 4 * i + 1) * LDS_STRIDE + bkr] = (_Float16)breg[i].y;
                lds_b[buf][(bnc + 4 * i + 2) * LDS_STRIDE + bkr] = (_Float16)breg[i].z;
                lds_b[buf][(bnc + 4 * i + 3) * LDS_STRIDE + bkr] = (_Float16)breg[i].w;
            }
        }
    };

    load_stage(0);
    store_stage(0);

    for (int s = 0; s < nstage; ++s) {
        __syncthreads();
        const int buf = s & 1;

        // issue next stage's global loads so they overlap the WMMA chain
        if (s + 1 < nstage) load_stage(s + 1);
        if (s + 2 < nstage) {
            const int kp = (s + 2) << 6;
            __builtin_prefetch(A + (size_t)(m0 + ar) * lda + kp + ac, 0, 3);
            if (BTRANS)
                __builtin_prefetch(B + (size_t)(n0 + bnr) * ldb + kp + bkc, 0, 3);
            else
                __builtin_prefetch(B + (size_t)(kp + bkr) * ldb + n0 + bnc, 0, 3);
        }

#pragma unroll
        for (int ks = 0; ks < 64; ks += 32) {
            // A fragments: lane<16 row: K ks+0..15
            //              lane>=16 row: K ks+16..31
            v16h afrag[2];
#pragma unroll
            for (int ai = 0; ai < 2; ++ai) {
                const _Float16* ap =
                    &lds_a[buf][(wm + ai * 16 + lh) * LDS_STRIDE + ks];
                v8h a0 = *(const v8h*)(ap + (hi ? 16 : 0));
                v8h a1 = *(const v8h*)(ap + (hi ? 24 : 8));
#pragma unroll
                for (int i = 0; i < 8; ++i) {
                    afrag[ai][i] = a0[i]; afrag[ai][i + 8] = a1[i];
                }
            }
            // B fragments: lane<16 col holds K ks+0..15; lane>=16 K ks+16..31
            v16h bfrag[2];
#pragma unroll
            for (int bi = 0; bi < 2; ++bi) {
                const _Float16* bp =
                    &lds_b[buf][(wn + bi * 16 + lh) * LDS_STRIDE + ks + (hi ? 16 : 0)];
                v8h b0 = *(const v8h*)bp;
                v8h b1 = *(const v8h*)(bp + 8);
#pragma unroll
                for (int i = 0; i < 8; ++i) {
                    bfrag[bi][i] = b0[i]; bfrag[bi][i + 8] = b1[i];
                }
            }

#pragma unroll
            for (int ai = 0; ai < 2; ++ai)
#pragma unroll
                for (int bi = 0; bi < 2; ++bi)
                    acc[ai][bi] = __builtin_amdgcn_wmma_f32_16x16x32_f16(
                        false, afrag[ai], false, bfrag[bi],
                        (short)0, acc[ai][bi], false, false);
        }

        if (s + 1 < nstage) store_stage((s + 1) & 1);
    }

    // store: VGPR i -> row (+i) lanes 0-15 / (+8+i) lanes 16-31
#pragma unroll
    for (int ai = 0; ai < 2; ++ai) {
        const int row0 = m0 + wm + ai * 16 + (hi ? 8 : 0);
#pragma unroll
        for (int bi = 0; bi < 2; ++bi) {
            const v8f a = acc[ai][bi];
            const int col = n0 + wn + bi * 16 + lh;
            const float bv = bias ? bias[col] : 0.0f;
#pragma unroll
            for (int i = 0; i < 8; ++i) {
                float r = alpha * a[i] + bv;
                float* cp = C + (size_t)(row0 + i) * ldc + col;
                if (accumulate)    *cp += r;
                else if (NTSTORE)  __builtin_nontemporal_store(r, cp);
                else               *cp = r;
            }
        }
    }
}

// ---------------------------------------------------------------------------
// Elementwise / reduction kernels
// ---------------------------------------------------------------------------
__global__ void __launch_bounds__(256)
embed_kernel(const int* __restrict__ idx, const float* __restrict__ emb,
             float* __restrict__ tok, float* __restrict__ x)
{
    const int r = blockIdx.x;
    const int tkn = idx[r];
    for (int d = threadIdx.x; d < DDIM; d += 256) {
        float v = emb[(size_t)tkn * DDIM + d];
        tok[(size_t)r * DDIM + d] = v;
        x[(size_t)r * DDIM + d]   = v;
    }
}

__global__ void fill_zero(float* __restrict__ p, int n)
{
    for (int i = blockIdx.x * blockDim.x + threadIdx.x; i < n; i += gridDim.x * blockDim.x)
        p[i] = 0.0f;
}

__global__ void copy_kernel(const float* __restrict__ s, float* __restrict__ d, int n)
{
    for (int i = blockIdx.x * blockDim.x + threadIdx.x; i < n; i += gridDim.x * blockDim.x)
        d[i] = s[i];
}

__global__ void add3_kernel(float* __restrict__ x, const float* __restrict__ res,
                            const float* __restrict__ tok, int n)
{
    for (int i = blockIdx.x * blockDim.x + threadIdx.x; i < n; i += gridDim.x * blockDim.x)
        x[i] = x[i] + res[i] + tok[i];
}

__global__ void __launch_bounds__(256)
ln_kernel(const float* __restrict__ x, const float* __restrict__ w,
          const float* __restrict__ b, float* __restrict__ out)
{
    __shared__ float red[256];
    const int row = blockIdx.x;
    const float* xr = x + (size_t)row * DDIM;

    float s = 0.0f;
    for (int d = threadIdx.x; d < DDIM; d += 256) s += xr[d];
    red[threadIdx.x] = s; __syncthreads();
    for (int o = 128; o > 0; o >>= 1) {
        if (threadIdx.x < o) red[threadIdx.x] += red[threadIdx.x + o];
        __syncthreads();
    }
    const float mean = red[0] * (1.0f / DDIM);
    __syncthreads();

    float vs = 0.0f;
    for (int d = threadIdx.x; d < DDIM; d += 256) {
        float t = xr[d] - mean; vs += t * t;
    }
    red[threadIdx.x] = vs; __syncthreads();
    for (int o = 128; o > 0; o >>= 1) {
        if (threadIdx.x < o) red[threadIdx.x] += red[threadIdx.x + o];
        __syncthreads();
    }
    const float rstd = rsqrtf(red[0] * (1.0f / DDIM) + 1e-5f);

    float* orow = out + (size_t)row * DDIM;
    for (int d = threadIdx.x; d < DDIM; d += 256)
        orow[d] = (xr[d] - mean) * rstd * w[d] + b[d];
}

// sLSTM gate math: Z row = [i | f | c | o], stab = max(max_d i, max_d f)
__global__ void __launch_bounds__(256)
slstm_gates(const float* __restrict__ Z, const float* __restrict__ h,
            const float* __restrict__ fb, float* __restrict__ hid)
{
    __shared__ float red[256];
    const int row = blockIdx.x;
    const float* zi = Z + (size_t)row * 4 * DDIM;
    const float* zf = zi + DDIM;
    const float* zc = zi + 2 * DDIM;
    const float* zo = zi + 3 * DDIM;

    float mx = -3.4e38f;
    for (int d = threadIdx.x; d < DDIM; d += 256)
        mx = fmaxf(mx, fmaxf(zi[d], zf[d]));
    red[threadIdx.x] = mx; __syncthreads();
    for (int o = 128; o > 0; o >>= 1) {
        if (threadIdx.x < o) red[threadIdx.x] = fmaxf(red[threadIdx.x], red[threadIdx.x + o]);
        __syncthreads();
    }
    const float stab = red[0];

    const float* hr = h + (size_t)row * DDIM;
    float* ho = hid + (size_t)row * DDIM;
    for (int d = threadIdx.x; d < DDIM; d += 256) {
        float iv = __expf(zi[d] - stab);
        float fv = __expf(zf[d] + fb[d] - stab);
        float cell = (fv * hr[d] + iv * tanhf(zc[d])) / (fv + iv);
        float og = 1.0f / (1.0f + __expf(-zo[d]));
        ho[d] = og * tanhf(cell);
    }
}

// row softmax over length TDIM with pre-scale
__global__ void __launch_bounds__(256)
softmax_kernel(float* __restrict__ S, float scale)
{
    __shared__ float red[256];
    const int row = blockIdx.x;
    float* sr = S + (size_t)row * TDIM;

    float mx = -3.4e38f;
    for (int d = threadIdx.x; d < TDIM; d += 256) mx = fmaxf(mx, sr[d] * scale);
    red[threadIdx.x] = mx; __syncthreads();
    for (int o = 128; o > 0; o >>= 1) {
        if (threadIdx.x < o) red[threadIdx.x] = fmaxf(red[threadIdx.x], red[threadIdx.x + o]);
        __syncthreads();
    }
    const float m = red[0];
    __syncthreads();

    float s = 0.0f;
    for (int d = threadIdx.x; d < TDIM; d += 256) {
        float e = __expf(sr[d] * scale - m);
        sr[d] = e; s += e;
    }
    red[threadIdx.x] = s; __syncthreads();
    for (int o = 128; o > 0; o >>= 1) {
        if (threadIdx.x < o) red[threadIdx.x] += red[threadIdx.x + o];
        __syncthreads();
    }
    const float inv = 1.0f / red[0];
    for (int d = threadIdx.x; d < TDIM; d += 256) sr[d] *= inv;
}

// mem = (sigmoid(fp+fb) * mem + exp(ip) * v) / (sigmoid(fp+fb) + exp(ip))
__global__ void mlstm_combine(float* __restrict__ mem, const float* __restrict__ v,
                              const float* __restrict__ fp, const float* __restrict__ ip,
                              const float* __restrict__ fb, int n)
{
    for (int i = blockIdx.x * blockDim.x + threadIdx.x; i < n; i += gridDim.x * blockDim.x) {
        int d = i % DDIM;
        float f  = 1.0f / (1.0f + __expf(-(fp[i] + fb[d])));
        float iv = __expf(ip[i]);
        mem[i] = (f * mem[i] + iv * v[i]) / (f + iv);
    }
}

// ---------------------------------------------------------------------------
// Orchestration
// ---------------------------------------------------------------------------
extern "C" void kernel_launch(void* const* d_in, const int* in_sizes, int n_in,
                              void* d_out, int out_size, void* d_ws, size_t ws_size,
                              hipStream_t stream)
{
    (void)in_sizes; (void)n_in; (void)out_size; (void)ws_size;

    const int*   idx  = (const int*)  d_in[0];
    const float* emb  = (const float*)d_in[1];
    const float* sW   = (const float*)d_in[2];
    const float* sWb  = (const float*)d_in[3];
    const float* sU   = (const float*)d_in[4];
    const float* sUb  = (const float*)d_in[5];
    const float* sO   = (const float*)d_in[6];
    const float* sOb  = (const float*)d_in[7];
    const float* sF   = (const float*)d_in[8];
    const float* sLNw = (const float*)d_in[9];
    const float* sLNb = (const float*)d_in[10];
    const float* mW   = (const float*)d_in[11];
    const float* mWb  = (const float*)d_in[12];
    const float* mO   = (const float*)d_in[13];
    const float* mOb  = (const float*)d_in[14];
    const float* mF   = (const float*)d_in[15];
    const float* mLNw = (const float*)d_in[16];
    const float* mLNb = (const float*)d_in[17];
    const float* lnfw = (const float*)d_in[18];
    const float* lnfb = (const float*)d_in[19];
    const float* head = (const float*)d_in[20];
    float* out = (float*)d_out;

    float* ws = (float*)d_ws;
    size_t off = 0;
    auto alloc = [&](size_t n) { float* p = ws + off; off += n; return p; };
    float* tok  = alloc((size_t)BT * DDIM);
    float* x    = alloc((size_t)BT * DDIM);
    float* xn   = alloc((size_t)BT * DDIM);
    float* res  = alloc((size_t)BT * DDIM);
    float* h    = alloc((size_t)BT * DDIM);
    float* hid  = alloc((size_t)BT * DDIM);
    float* Z    = alloc((size_t)BT * 4 * DDIM);
    float* proj = alloc((size_t)5 * BT * DDIM);
    float* sc   = alloc((size_t)BDIM * TDIM * TDIM);
    float* mem  = alloc((size_t)BT * DDIM);

    const dim3 blk(256);
    const float inv_sqrt_d = 0.03608439182435161f; // 1/sqrt(768)

    embed_kernel<<<BT, blk, 0, stream>>>(idx, emb, tok, x);

    for (int l = 0; l < LNUM; ++l) {
        copy_kernel<<<512, blk, 0, stream>>>(x, res, BT * DDIM);
        fill_zero<<<512, blk, 0, stream>>>(h, BT * DDIM);

        // ---- sLSTM sublayers ------------------------------------------------
        for (int s = 0; s < NS; ++s) {
            const int ps = l * NS + s;
            ln_kernel<<<BT, blk, 0, stream>>>(x, sLNw + (size_t)ps * DDIM,
                                              sLNb + (size_t)ps * DDIM, xn);
            // Z = xn @ W + Wb
            gemm_wmma_f16<false, false><<<dim3(4 * DDIM / 128, BT / 64), blk, 0, stream>>>(
                xn, sW + (size_t)ps * DDIM * 4 * DDIM, sWb + (size_t)ps * 4 * DDIM, Z,
                BT, 4 * DDIM, DDIM, DDIM, 4 * DDIM, 4 * DDIM, 1.0f, 0);
            // Z += h @ U + Ub
            gemm_wmma_f16<false, false><<<dim3(4 * DDIM / 128, BT / 64), blk, 0, stream>>>(
                h, sU + (size_t)ps * DDIM * 4 * DDIM, sUb + (size_t)ps * 4 * DDIM, Z,
                BT, 4 * DDIM, DDIM, DDIM, 4 * DDIM, 4 * DDIM, 1.0f, 1);
            slstm_gates<<<BT, blk, 0, stream>>>(Z, h, sF + (size_t)ps * DDIM, hid);
            // h_new = hid @ O + Ob
            gemm_wmma_f16<false, false><<<dim3(DDIM / 128, BT / 64), blk, 0, stream>>>(
                hid, sO + (size_t)ps * DDIM * DDIM, sOb + (size_t)ps * DDIM, h,
                BT, DDIM, DDIM, DDIM, DDIM, DDIM, 1.0f, 0);
            float* tswap = x; x = xn; xn = tswap;   // new x = LN'd old x
        }

        // ---- mLSTM sublayers ------------------------------------------------
        for (int m = 0; m < NM; ++m) {
            const int pm = l * NM + m;
            ln_kernel<<<BT, blk, 0, stream>>>(x, mLNw + (size_t)pm * DDIM,
                                              mLNb + (size_t)pm * DDIM, xn);
            for (int p = 0; p < 5; ++p) {
                gemm_wmma_f16<false, false><<<dim3(DDIM / 128, BT / 64), blk, 0, stream>>>(
                    xn, mW + ((size_t)pm * 5 + p) * DDIM * DDIM,
                    mWb + ((size_t)pm * 5 + p) * DDIM,
                    proj + (size_t)p * BT * DDIM,
                    BT, DDIM, DDIM, DDIM, DDIM, DDIM, 1.0f, 0);
            }
            const float* q  = proj;
            const float* k  = proj + (size_t)BT * DDIM;
            const float* v  = proj + (size_t)2 * BT * DDIM;
            const float* fp = proj + (size_t)3 * BT * DDIM;
            const float* ip = proj + (size_t)4 * BT * DDIM;

            for (int b = 0; b < BDIM; ++b) {
                // scores = q @ k^T  (NT GEMM)
                gemm_wmma_f16<true, false><<<dim3(TDIM / 128, TDIM / 64), blk, 0, stream>>>(
                    q + (size_t)b * TDIM * DDIM, k + (size_t)b * TDIM * DDIM, nullptr,
                    sc + (size_t)b * TDIM * TDIM,
                    TDIM, TDIM, DDIM, DDIM, DDIM, TDIM, 1.0f, 0);
            }
            softmax_kernel<<<BDIM * TDIM, blk, 0, stream>>>(sc, inv_sqrt_d);
            for (int b = 0; b < BDIM; ++b) {
                // mem = P @ v
                gemm_wmma_f16<false, false><<<dim3(DDIM / 128, TDIM / 64), blk, 0, stream>>>(
                    sc + (size_t)b * TDIM * TDIM, v + (size_t)b * TDIM * DDIM, nullptr,
                    mem + (size_t)b * TDIM * DDIM,
                    TDIM, DDIM, TDIM, TDIM, DDIM, DDIM, 1.0f, 0);
            }
            mlstm_combine<<<512, blk, 0, stream>>>(mem, v, fp, ip,
                                                   mF + (size_t)pm * DDIM, BT * DDIM);
            // x_new = mem @ O + Ob  (written into free xn buffer, then swap)
            gemm_wmma_f16<false, false><<<dim3(DDIM / 128, BT / 64), blk, 0, stream>>>(
                mem, mO + (size_t)pm * DDIM * DDIM, mOb + (size_t)pm * DDIM, xn,
                BT, DDIM, DDIM, DDIM, DDIM, DDIM, 1.0f, 0);
            float* tswap = x; x = xn; xn = tswap;
        }

        add3_kernel<<<512, blk, 0, stream>>>(x, res, tok, BT * DDIM);
    }

    // final LN + head projection -> logits [B,T,V] (streamed, non-temporal C)
    ln_kernel<<<BT, blk, 0, stream>>>(x, lnfw, lnfb, xn);
    gemm_wmma_f16<false, true><<<dim3(VDIM / 128, BT / 64), blk, 0, stream>>>(
        xn, head, nullptr, out, BT, VDIM, DDIM, DDIM, VDIM, VDIM, 1.0f, 0);
}
